// GNNGraphEncoder_19808389169449
// MI455X (gfx1250) — matrix operand order repaired and checked by reading, not compiled, not run on hardware
//
#include <hip/hip_runtime.h>
#include <hip/hip_bf16.h>
#include <math.h>

#define NNODES 50000
#define NEDGES 640000
#define HD     128
#define EMBD   500
#define NGRAPH 64
#define NLAYER 3

typedef __attribute__((ext_vector_type(16))) __bf16 v16bf;
typedef __attribute__((ext_vector_type(8)))  float  v8f;
typedef __attribute__((ext_vector_type(4)))  unsigned int u32x4;

union V16BF { v16bf v; u32x4 q[2]; };

static __device__ __forceinline__ unsigned short f32_to_bf16_bits(float f) {
    unsigned u = __float_as_uint(f);
    unsigned r = u + 0x7FFFu + ((u >> 16) & 1u);   // round-to-nearest-even
    return (unsigned short)(r >> 16);
}

// ---------------------------------------------------------------------------
// Pack a [128,128] f32 weight (K-major, W[k*128+n]) into per-lane WMMA-B
// layout: for tile t(0..7), kchunk c(0..3), lane l(0..31), elem e(0..15):
//   K = c*32 + (l>=16 ? 16 : 0) + e,  N = t*16 + (l&15)
// 16 contiguous bf16 (32B) per lane -> two b128 loads in the GEMM.
// ---------------------------------------------------------------------------
__global__ void pack_weight_bf16(const float* __restrict__ w,
                                 unsigned short* __restrict__ out) {
    int idx = blockIdx.x * blockDim.x + threadIdx.x;   // 0..16383
    if (idx >= 8 * 4 * 32 * 16) return;
    int e = idx & 15;
    int l = (idx >> 4) & 31;
    int c = (idx >> 9) & 3;
    int t = (idx >> 11) & 7;
    int K = c * 32 + ((l >> 4) << 4) + e;
    int n = t * 16 + (l & 15);
    out[idx] = f32_to_bf16_bits(w[K * HD + n]);
}

// ---------------------------------------------------------------------------
// Fused GEMM: out = relu(X[nrows,128] @ W[128,128] + bias)
// X row-major bf16; W pre-packed. The packed 32KB weight block is staged into
// LDS once per workgroup with gfx1250 async LDS-DMA
// (global_load_async_to_lds_b128, ASYNCcnt), then all 8 waves feed their
// v_wmma_f32_16x16x32_bf16 pipes from ds_load_b128.
// NOTE: the LDS destination offset is derived via ptrtoint of &wlds[0] so the
// buffer escapes -- the asm "memory" clobber then legally covers the writes
// (without this the compiler folds the ds loads to undef).
// One wave -> 16 rows x 128 cols (8 acc tiles, 4 k-steps, 32 WMMAs).
// ---------------------------------------------------------------------------
template <bool WF32, bool WBF>
__global__ __launch_bounds__(256)
void gemm_bias_relu_wmma(const unsigned short* __restrict__ xb,
                         const unsigned short* __restrict__ wp,
                         const float* __restrict__ bias,
                         float* __restrict__ out_f32,
                         unsigned short* __restrict__ out_bf,
                         int nrows) {
    __shared__ unsigned short wlds[8 * 4 * 32 * 16];   // 32 KB packed B

    // ---- async stage packed weights global -> LDS (8 x 16B per thread) ----
    {
        // low 32 bits of the flat pointer == LDS byte offset; the ptrtoint
        // also captures wlds so the asm may write it.
        unsigned lds_base = (unsigned)(size_t)(void*)&wlds[0];
        #pragma unroll
        for (int i = 0; i < 8; ++i) {
            unsigned boff   = (threadIdx.x * 8u + (unsigned)i) * 16u;
            unsigned ldsoff = lds_base + boff;
            asm volatile("global_load_async_to_lds_b128 %0, %1, %2"
                         :: "v"(ldsoff), "v"(boff), "s"(wp)
                         : "memory");
        }
        asm volatile("s_wait_asynccnt 0x0" ::: "memory");
    }
    __syncthreads();

    const int wave   = threadIdx.x >> 5;
    const int lane   = threadIdx.x & 31;
    const int hi     = lane >> 4;
    const int lanelo = lane & 15;
    const int m0     = (blockIdx.x * 8 + wave) * 16;
    if (m0 >= nrows) return;                 // wave-uniform: EXEC stays all-ones

    int row = m0 + lanelo;
    if (row >= nrows) row = nrows - 1;       // defensive clamp (N%16==0 anyway)

    v8f acc[8] = {};
    const u32x4* xrow = (const u32x4*)(xb + (size_t)row * HD); // 16 x 16B per row

    #pragma unroll
    for (int c = 0; c < 4; ++c) {
        V16BF a;
        // A layout: lanes<16 hold K=c*32+0..15; lanes>=16 hold K=c*32+16..31
        a.q[0] = xrow[c * 4 + 2 * hi];
        a.q[1] = xrow[c * 4 + 2 * hi + 1];
        #pragma unroll
        for (int t = 0; t < 8; ++t) {
            V16BF b;
            const u32x4* bp =
                (const u32x4*)(wlds + (size_t)(((t * 4 + c) * 32 + lane) * 16));
            b.q[0] = bp[0];
            b.q[1] = bp[1];
            acc[t] = __builtin_amdgcn_wmma_f32_16x16x32_bf16(
                false, a.v, false, b.v, (short)0, acc[t], false, false);
        }
    }

    #pragma unroll
    for (int t = 0; t < 8; ++t) {
        const int col = t * 16 + lanelo;
        const float bv = bias[col];
        #pragma unroll
        for (int r = 0; r < 8; ++r) {
            const int orow = m0 + r + hi * 8;   // D layout: lanes>=16 -> M+8
            if (orow < nrows) {
                float v = acc[t][r] + bv;
                v = v > 0.f ? v : 0.f;
                if constexpr (WF32) out_f32[(size_t)orow * HD + col] = v;
                if constexpr (WBF)  out_bf [(size_t)orow * HD + col] = f32_to_bf16_bits(v);
            }
        }
    }
}

// ---------------------------------------------------------------------------
// agg[dst,:] += h[src,:]  (one thread = one edge x 4 features)
// ---------------------------------------------------------------------------
__global__ void scatter_add(const float* __restrict__ h,
                            const int* __restrict__ ei,
                            float* __restrict__ agg) {
    long long gid = (long long)blockIdx.x * blockDim.x + threadIdx.x;
    int e = (int)(gid >> 5);
    if (e >= NEDGES) return;
    int f   = ((int)gid & 31) * 4;
    int src = ei[e];
    int dst = ei[NEDGES + e];
    const float4 v = *(const float4*)(h + (size_t)src * HD + f);
    float* d = agg + (size_t)dst * HD + f;
    atomicAdd(d + 0, v.x); atomicAdd(d + 1, v.y);
    atomicAdd(d + 2, v.z); atomicAdd(d + 3, v.w);
}

__global__ void add_cvt_bf16(const float* __restrict__ a,
                             const float* __restrict__ b,
                             unsigned short* __restrict__ o, int n) {
    int i = blockIdx.x * blockDim.x + threadIdx.x;
    if (i < n) o[i] = f32_to_bf16_bits(a[i] + b[i]);
}

// g[n] = dot(g1[n,:], w2) + b2   (one wave per node, shfl reduce)
__global__ void gate_score(const float* __restrict__ g1,
                           const float* __restrict__ w2,
                           const float* __restrict__ b2,
                           float* __restrict__ g) {
    int wave = threadIdx.x >> 5, lane = threadIdx.x & 31;
    int node = blockIdx.x * 8 + wave;
    if (node >= NNODES) return;
    const float* row = g1 + (size_t)node * HD;
    float s = 0.f;
    #pragma unroll
    for (int k = 0; k < 4; ++k) s += row[lane + 32 * k] * w2[lane + 32 * k];
    for (int o = 16; o > 0; o >>= 1) s += __shfl_xor(s, o, 32);
    if (lane == 0) g[node] = s + b2[0];
}

__global__ void init_ms(float* m, float* s) {
    int i = threadIdx.x;
    if (i < NGRAPH) { m[i] = -INFINITY; s[i] = 0.f; }
}

static __device__ __forceinline__ void atomicMaxFloat(float* addr, float v) {
    if (v >= 0.f) atomicMax((int*)addr, __float_as_int(v));
    else          atomicMin((unsigned int*)addr, __float_as_uint(v));
}

__global__ void seg_max(const float* __restrict__ g,
                        const int* __restrict__ batch, float* __restrict__ m) {
    int i = blockIdx.x * blockDim.x + threadIdx.x;
    if (i < NNODES) atomicMaxFloat(&m[batch[i]], g[i]);
}

__global__ void seg_exp(float* __restrict__ g, const int* __restrict__ batch,
                        const float* __restrict__ m, float* __restrict__ s) {
    int i = blockIdx.x * blockDim.x + threadIdx.x;
    if (i < NNODES) {
        int b = batch[i];
        float e = expf(g[i] - m[b]);
        g[i] = e;
        atomicAdd(&s[b], e);
    }
}

// pooled[b,:] += (e[n]/s[b]) * h[n,:]
__global__ void pool_nodes(const float* __restrict__ e,
                           const float* __restrict__ s,
                           const int* __restrict__ batch,
                           const float* __restrict__ h,
                           float* __restrict__ pooled) {
    long long gid = (long long)blockIdx.x * blockDim.x + threadIdx.x;
    int node = (int)(gid >> 5);
    if (node >= NNODES) return;
    int f = ((int)gid & 31) * 4;
    int b = batch[node];
    float alpha = e[node] / s[b];
    const float4 v = *(const float4*)(h + (size_t)node * HD + f);
    float* d = pooled + (size_t)b * HD + f;
    atomicAdd(d + 0, alpha * v.x); atomicAdd(d + 1, alpha * v.y);
    atomicAdd(d + 2, alpha * v.z); atomicAdd(d + 3, alpha * v.w);
}

// out[g,j] = pooled[g,:] . proj_w[:,j] + proj_b[j]   (coalesced over j)
__global__ void proj_kernel(const float* __restrict__ pooled,
                            const float* __restrict__ w,
                            const float* __restrict__ b,
                            float* __restrict__ out) {
    int j = blockIdx.x * blockDim.x + threadIdx.x;
    if (j >= NGRAPH * EMBD) return;
    int gi = j / EMBD, col = j % EMBD;
    const float* p = pooled + (size_t)gi * HD;
    float acc = b[col];
    #pragma unroll 4
    for (int k = 0; k < HD; ++k) acc += p[k] * w[k * EMBD + col];
    out[j] = acc;
}

// ---------------------------------------------------------------------------
extern "C" void kernel_launch(void* const* d_in, const int* in_sizes, int n_in,
                              void* d_out, int out_size, void* d_ws, size_t ws_size,
                              hipStream_t stream) {
    const float* x   = (const float*)d_in[0];            // [N,128]
    const int*   ei  = (const int*)d_in[1];              // [2,E]
    const int*   bat = (const int*)d_in[2];              // [N]
    // params: conv{l}_w1, conv{l}_b1, conv{l}_w2, conv{l}_b2 (l=0..2),
    //         gate_w1, gate_b1, gate_w2, gate_b2, proj_w, proj_b
    const float* cw[6]; const float* cb[6];
    for (int l = 0; l < NLAYER; ++l) {
        cw[2*l]   = (const float*)d_in[3 + 4*l + 0];
        cb[2*l]   = (const float*)d_in[3 + 4*l + 1];
        cw[2*l+1] = (const float*)d_in[3 + 4*l + 2];
        cb[2*l+1] = (const float*)d_in[3 + 4*l + 3];
    }
    const float* gate_w1 = (const float*)d_in[15];
    const float* gate_b1 = (const float*)d_in[16];
    const float* gate_w2 = (const float*)d_in[17];
    const float* gate_b2 = (const float*)d_in[18];
    const float* proj_w  = (const float*)d_in[19];
    const float* proj_b  = (const float*)d_in[20];
    float* out = (float*)d_out;

    // ---- workspace carve-up (256B aligned slices) ----
    char* ws = (char*)d_ws;
    size_t off = 0;
    auto carve = [&](size_t bytes) {
        char* p = ws + off;
        off += (bytes + 255) & ~(size_t)255;
        return p;
    };
    const size_t NH = (size_t)NNODES * HD;
    float*          h_buf  = (float*)carve(NH * sizeof(float));
    float*          agg    = (float*)carve(NH * sizeof(float));   // also g1 f32
    unsigned short* bfA    = (unsigned short*)carve(NH * sizeof(unsigned short));
    unsigned short* bfB    = (unsigned short*)carve(NH * sizeof(unsigned short));
    unsigned short* wpk    = (unsigned short*)carve(7 * 16384 * sizeof(unsigned short));
    float*          gsc    = (float*)carve(NNODES * sizeof(float));
    float*          gmax   = (float*)carve(NGRAPH * sizeof(float));
    float*          gsum   = (float*)carve(NGRAPH * sizeof(float));
    float*          pooled = (float*)carve((size_t)NGRAPH * HD * sizeof(float));
    (void)ws_size; (void)n_in; (void)in_sizes; (void)out_size;

    // ---- pack all 128x128 weights to WMMA-B bf16 layout ----
    for (int i = 0; i < 6; ++i)
        pack_weight_bf16<<<64, 256, 0, stream>>>(cw[i], wpk + (size_t)i * 16384);
    pack_weight_bf16<<<64, 256, 0, stream>>>(gate_w1, wpk + (size_t)6 * 16384);

    const int gemm_blocks = (NNODES + 127) / 128;   // 8 waves x 16 rows
    const int ew_blocks   = (NEDGES * 32 + 255) / 256;
    const int nh_blocks   = ((int)NH + 255) / 256;
    const int nw_blocks   = (NNODES + 7) / 8;
    const int n_blocks    = (NNODES + 255) / 256;
    const int pool_blocks = (NNODES * 32 + 255) / 256;

    const float* h_cur = x;
    for (int l = 0; l < NLAYER; ++l) {
        hipMemsetAsync(agg, 0, NH * sizeof(float), stream);
        scatter_add<<<ew_blocks, 256, 0, stream>>>(h_cur, ei, agg);
        add_cvt_bf16<<<nh_blocks, 256, 0, stream>>>(h_cur, agg, bfA, (int)NH);
        // z = relu((h+agg) @ W1 + b1)  -> bf16 only
        gemm_bias_relu_wmma<false, true><<<gemm_blocks, 256, 0, stream>>>(
            bfA, wpk + (size_t)(2 * l) * 16384, cb[2 * l], nullptr, bfB, NNODES);
        // h = relu(z @ W2 + b2)        -> f32 (+bf16 for gate after last layer)
        gemm_bias_relu_wmma<true, true><<<gemm_blocks, 256, 0, stream>>>(
            bfB, wpk + (size_t)(2 * l + 1) * 16384, cb[2 * l + 1], h_buf, bfA, NNODES);
        h_cur = h_buf;
    }

    // gate MLP layer 1: g1 = relu(h @ gate_w1 + gate_b1)  (f32 into agg)
    gemm_bias_relu_wmma<true, false><<<gemm_blocks, 256, 0, stream>>>(
        bfA, wpk + (size_t)6 * 16384, gate_b1, agg, nullptr, NNODES);
    // gate scores
    gate_score<<<nw_blocks, 256, 0, stream>>>(agg, gate_w2, gate_b2, gsc);

    // segment softmax + weighted pooling
    init_ms<<<1, 64, 0, stream>>>(gmax, gsum);
    seg_max<<<n_blocks, 256, 0, stream>>>(gsc, bat, gmax);
    seg_exp<<<n_blocks, 256, 0, stream>>>(gsc, bat, gmax, gsum);
    hipMemsetAsync(pooled, 0, (size_t)NGRAPH * HD * sizeof(float), stream);
    pool_nodes<<<pool_blocks, 256, 0, stream>>>(gsc, gsum, bat, h_buf, pooled);

    // projection [64,128]x[128,500]
    proj_kernel<<<(NGRAPH * EMBD + 255) / 256, 256, 0, stream>>>(
        pooled, proj_w, proj_b, out);
}
